// EncMtanRnn_10866267259561
// MI455X (gfx1250) — matrix operand order repaired and checked by reading, not compiled, not run on hardware
//
#include <hip/hip_runtime.h>
#include <hip/hip_bf16.h>

// ---------------------------------------------------------------------------
// mTAN encoder for MI455X (gfx1250): all heavy GEMMs on v_wmma_f32_16x16x32_bf16
// ---------------------------------------------------------------------------

typedef __bf16 bf16_t;
typedef __attribute__((ext_vector_type(16))) __bf16 v16bf;
typedef __attribute__((ext_vector_type(8)))  __bf16 v8bf;
typedef __attribute__((ext_vector_type(8)))  float  v8f;

#define B_   64
#define T_   512
#define Q_   256
#define DV_  128
#define ET_  128
#define NH_  8
#define HID_ 256

// sched_group_barrier masks
#define SG_WMMA   0x008
#define SG_VMEMRD 0x020
#define SG_DSRD   0x100
#define SGB(m, n) __builtin_amdgcn_sched_group_barrier((m), (n), 0)

// Assemble the 16-element bf16 A/B fragment from two 16-byte chunks
static __device__ __forceinline__ v16bf load16(const bf16_t* lo, const bf16_t* hi) {
  v8bf a = *(const v8bf*)lo;
  v8bf b = *(const v8bf*)hi;
  return __builtin_shufflevector(a, b, 0,1,2,3,4,5,6,7,8,9,10,11,12,13,14,15);
}

// ------------------------- small VALU helper kernels -----------------------

__global__ void cvt_bf16_kernel(const float* __restrict__ src, bf16_t* __restrict__ dst, int n) {
  int i = blockIdx.x * blockDim.x + threadIdx.x;
  if (i < n) dst[i] = (bf16_t)src[i];
}

// Pack GRU recurrent weights: Wcat[k][0:512)=Wh, [512:768)=Whn  (bf16)
__global__ void pack_wcat_kernel(const float* __restrict__ Wh, const float* __restrict__ Whn,
                                 bf16_t* __restrict__ dst) {
  int i = blockIdx.x * blockDim.x + threadIdx.x;
  if (i >= 256 * 768) return;
  int k = i / 768, j = i % 768;
  float v = (j < 512) ? Wh[k * 512 + j] : Whn[k * 256 + (j - 512)];
  dst[i] = (bf16_t)v;
}

// Sinusoidal time embedding -> bf16 rows [rows x 128]
__global__ void embed_kernel(const float* __restrict__ pos, bf16_t* __restrict__ out, int rows) {
  int r = blockIdx.x * blockDim.x + threadIdx.x;
  if (r >= rows) return;
  float p = 48.0f * pos[r];
  bf16_t* o = out + (size_t)r * ET_;
  const float cexp = -2.302585093f / 64.0f;  // -ln(10)/64
  for (int j = 0; j < 64; ++j) {
    float arg = p * __expf(cexp * (float)j);
    o[2 * j]     = (bf16_t)__sinf(arg);
    o[2 * j + 1] = (bf16_t)__cosf(arg);
  }
}

// kproj[B*T,128] -> kT[b][h][d(0..31, zero-padded)][t]  bf16
__global__ void reorder_k_kernel(const float* __restrict__ kproj, bf16_t* __restrict__ kT) {
  int i = blockIdx.x * blockDim.x + threadIdx.x;
  if (i >= B_ * NH_ * 32 * T_) return;
  int t = i & (T_ - 1);
  int d = (i >> 9) & 31;
  int h = (i >> 14) & 7;
  int b = i >> 17;
  float v = (d < 16) ? kproj[((size_t)(b * T_ + t)) * ET_ + h * 16 + d] : 0.0f;
  kT[i] = (bf16_t)v;
}

// qproj[Q,128] -> qpad[h][q][d(0..31, zero-padded)] bf16, scaled by 1/sqrt(dk)
__global__ void reorder_q_kernel(const float* __restrict__ qproj, bf16_t* __restrict__ qpad) {
  int i = blockIdx.x * blockDim.x + threadIdx.x;
  if (i >= NH_ * Q_ * 32) return;
  int d  = i & 31;
  int qi = (i >> 5) & 255;
  int h  = i >> 13;
  float v = (d < 16) ? 0.25f * qproj[(size_t)qi * ET_ + h * 16 + d] : 0.0f;
  qpad[i] = (bf16_t)v;
}

// ------------------------- generic bf16 WMMA GEMM --------------------------
// C[M,N] = A[M,K](bf16,row-major) @ W[K,N](bf16,row-major) + bias
// block = 128 threads (4 waves stacked over M), each wave does 16 rows x 64 cols.
// sched_group_barrier pins: one big VMEM-read clause, then 4 back-to-back WMMAs.
__global__ void gemm_bf16_kernel(const bf16_t* __restrict__ A, const bf16_t* __restrict__ W,
                                 const float* __restrict__ bias,
                                 float* __restrict__ outF, bf16_t* __restrict__ outB,
                                 int M, int N, int K) {
  int lane = threadIdx.x & 31;
  int wave = threadIdx.x >> 5;
  int m0 = blockIdx.y * 64 + wave * 16;
  int n0 = blockIdx.x * 64;
  int m = lane & 15, hl = lane >> 4, kb = hl * 8;
  v8f acc[4] = {};
  const bf16_t* arow = A + (size_t)(m0 + m) * K;
  for (int k0 = 0; k0 < K; k0 += 32) {
    const bf16_t* wrow = W + (size_t)k0 * N + n0 + m;
    // unconditional speculative prefetch of the next k-step (dropped if OOB)
    __builtin_prefetch(arow + k0 + 32, 0, 1);
    __builtin_prefetch(W + (size_t)(k0 + 32 + lane) * N + n0, 0, 1);
    v16bf a = load16(arow + k0 + kb, arow + k0 + 16 + kb);
    v16bf bv[4];
#pragma unroll
    for (int t = 0; t < 4; ++t) for (int s = 0; s < 16; ++s) bv[t][s] = wrow[(size_t)(kb + (s & 7) + ((s >> 3) << 4)) * N + t * 16];
#pragma unroll
    for (int t = 0; t < 4; ++t)
      acc[t] = __builtin_amdgcn_wmma_f32_16x16x32_bf16(false, a, false, bv[t],
                                                       (short)0, acc[t], false, false);
    SGB(SG_VMEMRD, 16);  // all loads (+prefetches) first
    SGB(SG_WMMA, 4);     // then the 4 WMMAs together
  }
#pragma unroll
  for (int t = 0; t < 4; ++t) {
    int col = n0 + t * 16 + (lane & 15);
    float bvv = bias ? bias[col] : 0.0f;
#pragma unroll
    for (int r = 0; r < 8; ++r) {
      int row = m0 + r + 8 * hl;
      float v = acc[t][r] + bvv;
      if (outF) outF[(size_t)row * N + col] = v;
      if (outB) outB[(size_t)row * N + col] = (bf16_t)v;
    }
  }
}

// ---------------------- fused attention (per b,h,q-tile) -------------------
// one wave per block: scores (WMMA, dk padded to 32) -> softmax -> p @ x (WMMA)
__global__ void attn_kernel(const bf16_t* __restrict__ qpad,  // [NH][Q][32]
                            const bf16_t* __restrict__ kT,    // [B][NH][32][T]
                            const bf16_t* __restrict__ xbf,   // [B][T][DV]
                            bf16_t* __restrict__ attn)        // [B][Q][NH*DV]
{
  __shared__ __attribute__((aligned(32))) float  S[16 * T_];
  __shared__ __attribute__((aligned(32))) bf16_t P[16 * T_];
  int bid = blockIdx.x;
  int qt = bid & 15;
  int h  = (bid >> 4) & 7;
  int b  = bid >> 7;
  int lane = threadIdx.x;
  int m = lane & 15, hl = lane >> 4, kb = hl * 8;

  // A fragment: q tile 16x32 (upper 16 K are zero padding)
  const bf16_t* qrow = qpad + ((size_t)h * Q_ + qt * 16 + m) * 32;
  v16bf aq = load16(qrow + kb, qrow + 16 + kb);

  // scores S[16 x 512] = q @ k^T   (B fragment: lane column n=lane%16, elements over K in table order)
  const bf16_t* krow = kT + (((size_t)b * NH_ + h) * 32) * T_ + m;
  for (int tt = 0; tt < T_ / 16; tt += 2) {
    v16bf bk0, bk1;
    for (int s = 0; s < 16; ++s) {
      size_t ko = (size_t)(kb + (s & 7) + ((s >> 3) << 4)) * T_ + tt * 16;
      bk0[s] = krow[ko];
      bk1[s] = krow[ko + 16];
    }
    v8f c0 = {}, c1 = {};
    c0 = __builtin_amdgcn_wmma_f32_16x16x32_bf16(false, aq, false, bk0, (short)0, c0, false, false);
    c1 = __builtin_amdgcn_wmma_f32_16x16x32_bf16(false, aq, false, bk1, (short)0, c1, false, false);
    int col = tt * 16 + (lane & 15);
#pragma unroll
    for (int r = 0; r < 8; ++r) {
      S[(r + 8 * hl) * T_ + col]      = c0[r];
      S[(r + 8 * hl) * T_ + col + 16] = c1[r];
    }
    SGB(SG_VMEMRD, 4);
    SGB(SG_WMMA, 2);
  }
  __syncthreads();

  // softmax over T: lane handles row m, half hl covers 256 cols; combine halves
  {
    const float* srow = S + m * T_ + hl * 256;
    float mx = -1e30f;
    for (int t = 0; t < 256; ++t) mx = fmaxf(mx, srow[t]);
    mx = fmaxf(mx, __shfl_xor(mx, 16));
    float sum = 0.0f;
    for (int t = 0; t < 256; ++t) sum += __expf(srow[t] - mx);
    sum += __shfl_xor(sum, 16);
    float inv = 1.0f / sum;
    bf16_t* prow = P + m * T_ + hl * 256;
    for (int t = 0; t < 256; ++t) prow[t] = (bf16_t)(__expf(srow[t] - mx) * inv);
  }
  __syncthreads();

  // attn tile = P[16 x 512] @ x_b[512 x 128]; batch loads, then 8 WMMAs
  v8f acc[8] = {};
  for (int k0 = 0; k0 < T_; k0 += 32) {
    v16bf ap = load16(&P[m * T_ + k0 + kb], &P[m * T_ + k0 + 16 + kb]);
    const bf16_t* xrow = xbf + ((size_t)b * T_ + k0) * DV_ + m;
    __builtin_prefetch(xbf + ((size_t)b * T_ + k0 + 32 + lane) * DV_, 0, 1);
    v16bf bx[8];
#pragma unroll
    for (int nt = 0; nt < 8; ++nt) for (int s = 0; s < 16; ++s) bx[nt][s] = xrow[(size_t)(kb + (s & 7) + ((s >> 3) << 4)) * DV_ + nt * 16];
#pragma unroll
    for (int nt = 0; nt < 8; ++nt)
      acc[nt] = __builtin_amdgcn_wmma_f32_16x16x32_bf16(false, ap, false, bx[nt],
                                                        (short)0, acc[nt], false, false);
    SGB(SG_DSRD, 2);     // A fragment from LDS
    SGB(SG_VMEMRD, 18);  // x tiles (+prefetch)
    SGB(SG_WMMA, 8);
  }
#pragma unroll
  for (int nt = 0; nt < 8; ++nt) {
    int col = h * DV_ + nt * 16 + (lane & 15);
#pragma unroll
    for (int r = 0; r < 8; ++r) {
      int qi = qt * 16 + r + 8 * hl;
      attn[((size_t)b * Q_ + qi) * (NH_ * DV_) + col] = (bf16_t)acc[nt][r];
    }
  }
}

// ------------------------- persistent bidirectional GRU --------------------
// grid = 8 blocks: dir(2) x batch-tile(4, 16 rows each); block = 256 threads (8 waves)
// each step: gates[16x768] = h[16x256] @ Wcat[256x768] (WMMA) then VALU gate math
__global__ void gru_kernel(const float* __restrict__ gif, const float* __restrict__ gib,
                           const bf16_t* __restrict__ Wf, const bf16_t* __restrict__ Wb,
                           const float* __restrict__ bhn_f, const float* __restrict__ bhn_b,
                           float* __restrict__ hf, float* __restrict__ hb) {
  __shared__ __attribute__((aligned(32))) bf16_t hsh[16 * HID_];
  __shared__ __attribute__((aligned(32))) float  gates[16 * 768];
  int dir = blockIdx.x >> 2;
  int b0  = (blockIdx.x & 3) * 16;
  const float*  gi  = dir ? gib   : gif;
  const bf16_t* W   = dir ? Wb    : Wf;
  const float*  bhn = dir ? bhn_b : bhn_f;
  float*        out = dir ? hb    : hf;

  int tid = threadIdx.x;
  int wave = tid >> 5, lane = tid & 31;
  int m = lane & 15, hl = lane >> 4, kb = hl * 8;

  for (int i = tid; i < 16 * HID_; i += 256) hsh[i] = (bf16_t)0.0f;
  __syncthreads();

  for (int step = 0; step < Q_; ++step) {
    int q = dir ? (Q_ - 1 - step) : step;
    // phase 1: recurrent GEMM (wave handles 6 n-tiles = 96 cols)
    v8f acc[6] = {};
    for (int k0 = 0; k0 < HID_; k0 += 32) {
      v16bf ah = load16(&hsh[m * HID_ + k0 + kb], &hsh[m * HID_ + k0 + 16 + kb]);
      const bf16_t* wrow = W + (size_t)k0 * 768 + wave * 96 + m;
      v16bf bw[6];
#pragma unroll
      for (int t = 0; t < 6; ++t) for (int s = 0; s < 16; ++s) bw[t][s] = wrow[(size_t)(kb + (s & 7) + ((s >> 3) << 4)) * 768 + t * 16];
#pragma unroll
      for (int t = 0; t < 6; ++t)
        acc[t] = __builtin_amdgcn_wmma_f32_16x16x32_bf16(false, ah, false, bw[t],
                                                         (short)0, acc[t], false, false);
      SGB(SG_DSRD, 2);
      SGB(SG_VMEMRD, 12);
      SGB(SG_WMMA, 6);
    }
#pragma unroll
    for (int t = 0; t < 6; ++t) {
      int col = wave * 96 + t * 16 + (lane & 15);
#pragma unroll
      for (int r = 0; r < 8; ++r) gates[(r + 8 * hl) * 768 + col] = acc[t][r];
    }
    __syncthreads();
    // phase 2: gate math (each thread owns distinct (bi,j) elements)
    for (int idx = tid; idx < 16 * HID_; idx += 256) {
      int bi = idx >> 8;
      int j  = idx & 255;
      size_t rowg = (size_t)(b0 + bi) * Q_ + q;
      const float* g = gi + rowg * 768;
      float ir = g[j], iz = g[256 + j], inn = g[512 + j];
      float hr = gates[bi * 768 + j];
      float hz = gates[bi * 768 + 256 + j];
      float hn = gates[bi * 768 + 512 + j];
      float hp = (float)hsh[bi * HID_ + j];
      float r = 1.0f / (1.0f + __expf(-(ir + hr)));
      float z = 1.0f / (1.0f + __expf(-(iz + hz)));
      float n = tanhf(inn + r * (hn + bhn[j]));
      float hnew = (1.0f - z) * n + z * hp;
      out[rowg * HID_ + j] = hnew;
      hsh[bi * HID_ + j] = (bf16_t)hnew;
    }
    __syncthreads();
  }
}

// ------------------------- final MLP (small, VALU) -------------------------

__global__ void mlp1_kernel(const float* __restrict__ hf, const float* __restrict__ hb,
                            const float* __restrict__ W1, const float* __restrict__ b1,
                            float* __restrict__ mid) {
  __shared__ float catr[512];
  int row = blockIdx.x, tid = threadIdx.x;
  for (int i = tid; i < 256; i += 64) {
    catr[i]       = hf[(size_t)row * 256 + i];
    catr[256 + i] = hb[(size_t)row * 256 + i];
  }
  __syncthreads();
  for (int c = tid; c < 50; c += 64) {
    float s = b1[c];
    for (int k = 0; k < 512; ++k) s += catr[k] * W1[k * 50 + c];
    mid[(size_t)row * 50 + c] = fmaxf(s, 0.0f);
  }
}

__global__ void mlp2_kernel(const float* __restrict__ mid, const float* __restrict__ W2,
                            const float* __restrict__ b2, float* __restrict__ out) {
  __shared__ float mm[50];
  int row = blockIdx.x, tid = threadIdx.x;
  for (int i = tid; i < 50; i += 64) mm[i] = mid[(size_t)row * 50 + i];
  __syncthreads();
  if (tid < 64) {
    float s = b2[tid];
    for (int k = 0; k < 50; ++k) s += mm[k] * W2[k * 64 + tid];
    out[(size_t)row * 64 + tid] = s;
  }
}

// ---------------------------------------------------------------------------

extern "C" void kernel_launch(void* const* d_in, const int* in_sizes, int n_in,
                              void* d_out, int out_size, void* d_ws, size_t ws_size,
                              hipStream_t stream) {
  (void)in_sizes; (void)n_in; (void)out_size; (void)ws_size;
  const float* x      = (const float*)d_in[0];
  const float* tsteps = (const float*)d_in[1];
  const float* query  = (const float*)d_in[2];
  const float* Wq = (const float*)d_in[3];  const float* bq = (const float*)d_in[4];
  const float* Wk = (const float*)d_in[5];  const float* bk = (const float*)d_in[6];
  const float* Wo = (const float*)d_in[7];  const float* bo = (const float*)d_in[8];
  const float* gf_Wi = (const float*)d_in[9];   const float* gf_bi = (const float*)d_in[10];
  const float* gf_Wh = (const float*)d_in[11];  const float* gf_Whn = (const float*)d_in[12];
  const float* gf_bhn = (const float*)d_in[13];
  const float* gb_Wi = (const float*)d_in[14];  const float* gb_bi = (const float*)d_in[15];
  const float* gb_Wh = (const float*)d_in[16];  const float* gb_Whn = (const float*)d_in[17];
  const float* gb_bhn = (const float*)d_in[18];
  const float* W1 = (const float*)d_in[19]; const float* b1 = (const float*)d_in[20];
  const float* W2 = (const float*)d_in[21]; const float* b2 = (const float*)d_in[22];
  float* out = (float*)d_out;

  // carve workspace
  char* base = (char*)d_ws;
  size_t off = 0;
  auto carve = [&](size_t bytes) -> char* {
    char* p = base + off;
    off += (bytes + 255) & ~(size_t)255;
    return p;
  };
  const int MK = B_ * T_;        // 32768
  const int MR = B_ * Q_;        // 16384
  bf16_t* Wq_bf   = (bf16_t*)carve((size_t)128 * 128 * 2);
  bf16_t* Wk_bf   = (bf16_t*)carve((size_t)128 * 128 * 2);
  bf16_t* Wo_bf   = (bf16_t*)carve((size_t)1024 * 256 * 2);
  bf16_t* gfWi_bf = (bf16_t*)carve((size_t)256 * 768 * 2);
  bf16_t* gbWi_bf = (bf16_t*)carve((size_t)256 * 768 * 2);
  bf16_t* Wcat_f  = (bf16_t*)carve((size_t)256 * 768 * 2);
  bf16_t* Wcat_b  = (bf16_t*)carve((size_t)256 * 768 * 2);
  bf16_t* kemb_bf = (bf16_t*)carve((size_t)MK * ET_ * 2);
  bf16_t* qemb_bf = (bf16_t*)carve((size_t)Q_ * ET_ * 2);
  bf16_t* x_bf    = (bf16_t*)carve((size_t)MK * DV_ * 2);
  float*  kproj_f = (float*) carve((size_t)MK * ET_ * 4);
  float*  qproj_f = (float*) carve((size_t)Q_ * ET_ * 4);
  bf16_t* kT_bf   = (bf16_t*)carve((size_t)B_ * NH_ * 32 * T_ * 2);
  bf16_t* q_pad   = (bf16_t*)carve((size_t)NH_ * Q_ * 32 * 2);
  bf16_t* attn_bf = (bf16_t*)carve((size_t)MR * 1024 * 2);
  bf16_t* h_bf    = (bf16_t*)carve((size_t)MR * HID_ * 2);
  float*  gif     = (float*) carve((size_t)MR * 768 * 4);
  float*  gib     = (float*) carve((size_t)MR * 768 * 4);
  float*  hf      = (float*) carve((size_t)MR * HID_ * 4);
  float*  hb      = (float*) carve((size_t)MR * HID_ * 4);
  float*  mid     = (float*) carve((size_t)MR * 50 * 4);

  auto cdiv = [](int a, int b) { return (a + b - 1) / b; };

  // weight conversions / packing
  cvt_bf16_kernel<<<cdiv(128 * 128, 256), 256, 0, stream>>>(Wq, Wq_bf, 128 * 128);
  cvt_bf16_kernel<<<cdiv(128 * 128, 256), 256, 0, stream>>>(Wk, Wk_bf, 128 * 128);
  cvt_bf16_kernel<<<cdiv(1024 * 256, 256), 256, 0, stream>>>(Wo, Wo_bf, 1024 * 256);
  cvt_bf16_kernel<<<cdiv(256 * 768, 256), 256, 0, stream>>>(gf_Wi, gfWi_bf, 256 * 768);
  cvt_bf16_kernel<<<cdiv(256 * 768, 256), 256, 0, stream>>>(gb_Wi, gbWi_bf, 256 * 768);
  cvt_bf16_kernel<<<cdiv(MK * DV_, 256), 256, 0, stream>>>(x, x_bf, MK * DV_);
  pack_wcat_kernel<<<cdiv(256 * 768, 256), 256, 0, stream>>>(gf_Wh, gf_Whn, Wcat_f);
  pack_wcat_kernel<<<cdiv(256 * 768, 256), 256, 0, stream>>>(gb_Wh, gb_Whn, Wcat_b);

  // time embeddings
  embed_kernel<<<cdiv(MK, 256), 256, 0, stream>>>(tsteps, kemb_bf, MK);
  embed_kernel<<<cdiv(Q_, 256), 256, 0, stream>>>(query, qemb_bf, Q_);

  // k/q projections (WMMA)
  gemm_bf16_kernel<<<dim3(128 / 64, MK / 64), 128, 0, stream>>>(
      kemb_bf, Wk_bf, bk, kproj_f, nullptr, MK, 128, 128);
  gemm_bf16_kernel<<<dim3(128 / 64, Q_ / 64), 128, 0, stream>>>(
      qemb_bf, Wq_bf, bq, qproj_f, nullptr, Q_, 128, 128);

  reorder_k_kernel<<<cdiv(B_ * NH_ * 32 * T_, 256), 256, 0, stream>>>(kproj_f, kT_bf);
  reorder_q_kernel<<<cdiv(NH_ * Q_ * 32, 256), 256, 0, stream>>>(qproj_f, q_pad);

  // fused attention: B*NH*(Q/16) single-wave workgroups
  attn_kernel<<<B_ * NH_ * (Q_ / 16), 32, 0, stream>>>(q_pad, kT_bf, x_bf, attn_bf);

  // h = attn @ Wo + bo  (bf16 out, feeds GRU input gates)
  gemm_bf16_kernel<<<dim3(HID_ / 64, MR / 64), 128, 0, stream>>>(
      attn_bf, Wo_bf, bo, nullptr, h_bf, MR, HID_, 1024);

  // GRU input gates for both directions
  gemm_bf16_kernel<<<dim3(768 / 64, MR / 64), 128, 0, stream>>>(
      h_bf, gfWi_bf, gf_bi, gif, nullptr, MR, 768, HID_);
  gemm_bf16_kernel<<<dim3(768 / 64, MR / 64), 128, 0, stream>>>(
      h_bf, gbWi_bf, gb_bi, gib, nullptr, MR, 768, HID_);

  // persistent bidirectional GRU (2 dirs x 4 batch tiles)
  gru_kernel<<<8, 256, 0, stream>>>(gif, gib, Wcat_f, Wcat_b, gf_bhn, gb_bhn, hf, hb);

  // final MLP
  mlp1_kernel<<<MR, 64, 0, stream>>>(hf, hb, W1, b1, mid);
  mlp2_kernel<<<MR, 64, 0, stream>>>(mid, W2, b2, out);
}
